// ASAP_58033598104013
// MI455X (gfx1250) — matrix operand order repaired and checked by reading, not compiled, not run on hardware
//
#include <hip/hip_runtime.h>
#include <hip/hip_bf16.h>

typedef __attribute__((ext_vector_type(16))) _Float16 v16h;
typedef __attribute__((ext_vector_type(8)))  float    v8f;

#define NEGV  (-1000000000.0f)
#define ENCNEG 0x007FFFFFu   /* fenc(-inf) */

__device__ __forceinline__ unsigned fenc(float f){
  unsigned u = __float_as_uint(f);
  return (u & 0x80000000u) ? ~u : (u | 0x80000000u);
}
__device__ __forceinline__ float fdec(unsigned u){
  return (u & 0x80000000u) ? __uint_as_float(u & 0x7fffffffu) : __uint_as_float(~u);
}
__device__ __forceinline__ void edge_sd(int e, int E, const int* src, const int* dst, int& s, int& d){
  if (e < E){ s = src[e]; d = dst[e]; } else { s = e - E; d = e - E; }
}
__device__ __forceinline__ float lrelu02(float v){ return v >= 0.f ? v : 0.2f*v; }

// ---------------------------------------------------------------- fills
__global__ void k_fill_f32(float* p, long long n, float v){
  long long i = (long long)blockIdx.x*256 + threadIdx.x;
  if (i < n) p[i] = v;
}
__global__ void k_fill_u32(unsigned* p, long long n, unsigned v){
  long long i = (long long)blockIdx.x*256 + threadIdx.x;
  if (i < n) p[i] = v;
}

// ---------------------------------------------------------------- WMMA GEMM
__device__ __forceinline__ v8f wmma32(v16h a, v16h b, v8f c){
  return __builtin_amdgcn_wmma_f32_16x16x32_f16(false, a, false, b, (short)0, c, false, false);
}

// A fragment (16x32): lanes 0-15 row m hold K=j (j<16); lanes 16-31 hold K=16+j.
template<int TRANS>
__device__ __forceinline__ v16h load_a_fast(const float* __restrict__ A, int lda,
                                            int mBase, int k0, int half, int l16){
  v16h a;
  int m = mBase + l16;
#pragma unroll
  for (int j = 0; j < 16; ++j) {
    int ka = k0 + half*16 + j;
    float av = TRANS ? A[(size_t)ka*lda + m] : A[(size_t)m*lda + ka];
    a[j] = (_Float16)av;
  }
  return a;
}
template<int TRANS>
__device__ __forceinline__ v16h load_a_guard(const float* __restrict__ A, int lda,
                                             int mBase, int k0, int half, int l16,
                                             int M, int K){
  v16h a;
  int m  = mBase + l16;
  int mc = m < M ? m : (M - 1);
#pragma unroll
  for (int j = 0; j < 16; ++j) {
    int ka = k0 + half*16 + j;
    int kc = ka < K ? ka : (K - 1);
    float av = TRANS ? A[(size_t)kc*lda + mc] : A[(size_t)mc*lda + kc];
    a[j] = (m < M && ka < K) ? (_Float16)av : (_Float16)0.f;
  }
  return a;
}
// B fragment (32x16): lanes 0-15 column n hold K=j (j<16); lanes 16-31 hold K=16+j.
template<int TRANS>
__device__ __forceinline__ v16h load_b_fast(const float* __restrict__ B, int ldb,
                                            int nBase, int k0, int half, int l16){
  v16h b;
  int nn = nBase + l16;
#pragma unroll
  for (int j = 0; j < 16; ++j) {
    int kb = k0 + half*16 + j;
    float bv = TRANS ? B[(size_t)nn*ldb + kb] : B[(size_t)kb*ldb + nn];
    b[j] = (_Float16)bv;
  }
  return b;
}
template<int TRANS>
__device__ __forceinline__ v16h load_b_guard(const float* __restrict__ B, int ldb,
                                             int nBase, int k0, int half, int l16,
                                             int N, int K){
  v16h b;
  int nn = nBase + l16;
  int nc = nn < N ? nn : (N - 1);
#pragma unroll
  for (int j = 0; j < 16; ++j) {
    int kb = k0 + half*16 + j;
    int kc = kb < K ? kb : (K - 1);
    float bv = TRANS ? B[(size_t)nc*ldb + kc] : B[(size_t)kc*ldb + nc];
    b[j] = (nn < N && kb < K) ? (_Float16)bv : (_Float16)0.f;
  }
  return b;
}
__device__ __forceinline__ void store_tile(v8f c, const float* __restrict__ Cadd,
                                           const float* __restrict__ bias,
                                           float* __restrict__ D, int ldd,
                                           int m0, int n0, int M, int N,
                                           int half, int l16, int act){
#pragma unroll
  for (int r = 0; r < 8; ++r) {
    int m  = m0 + r + 8*half;
    int nn = n0 + l16;
    if (m < M && nn < N) {
      float v = c[r];
      if (Cadd) v += Cadd[(size_t)m*ldd + nn];
      if (bias) v += bias[nn];
      if (act == 1) v = v > 0.f ? v : 0.f;                  // relu
      else if (act == 2) v = v > 0.f ? v : (expf(v) - 1.f); // elu
      D[(size_t)m*ldd + nn] = v;
    }
  }
}

// D = act( opA(MxK) * opB(KxN) + Cadd + bias ), 32x32 tile per wave, 2x2 WMMA blocking.
template<int TA, int TB>
__global__ __launch_bounds__(32)
void k_wmma_gemm_t(const float* __restrict__ A, int lda,
                   const float* __restrict__ B, int ldb,
                   const float* __restrict__ Cadd, const float* __restrict__ bias,
                   float* __restrict__ D, int ldd,
                   int M, int N, int K, int act)
{
  int lane = threadIdx.x;
  int half = lane >> 4;
  int l16  = lane & 15;
  int m0 = blockIdx.y * 32;
  int n0 = blockIdx.x * 32;
  v8f c00 = {}, c01 = {}, c10 = {}, c11 = {};
  const bool interior = (m0 + 32 <= M) && (n0 + 32 <= N);  // uniform per block
  const int Kmain = K & ~31;
  if (interior) {
    for (int k0 = 0; k0 < Kmain; k0 += 32) {
      v16h a0 = load_a_fast<TA>(A, lda, m0,      k0, half, l16);
      v16h a1 = load_a_fast<TA>(A, lda, m0 + 16, k0, half, l16);
      v16h b0 = load_b_fast<TB>(B, ldb, n0,      k0, half, l16);
      v16h b1 = load_b_fast<TB>(B, ldb, n0 + 16, k0, half, l16);
      c00 = wmma32(a0, b0, c00);
      c01 = wmma32(a0, b1, c01);
      c10 = wmma32(a1, b0, c10);
      c11 = wmma32(a1, b1, c11);
    }
    if (Kmain < K) {
      v16h a0 = load_a_guard<TA>(A, lda, m0,      Kmain, half, l16, M, K);
      v16h a1 = load_a_guard<TA>(A, lda, m0 + 16, Kmain, half, l16, M, K);
      v16h b0 = load_b_guard<TB>(B, ldb, n0,      Kmain, half, l16, N, K);
      v16h b1 = load_b_guard<TB>(B, ldb, n0 + 16, Kmain, half, l16, N, K);
      c00 = wmma32(a0, b0, c00);
      c01 = wmma32(a0, b1, c01);
      c10 = wmma32(a1, b0, c10);
      c11 = wmma32(a1, b1, c11);
    }
  } else {
    for (int k0 = 0; k0 < K; k0 += 32) {
      v16h a0 = load_a_guard<TA>(A, lda, m0,      k0, half, l16, M, K);
      v16h a1 = load_a_guard<TA>(A, lda, m0 + 16, k0, half, l16, M, K);
      v16h b0 = load_b_guard<TB>(B, ldb, n0,      k0, half, l16, N, K);
      v16h b1 = load_b_guard<TB>(B, ldb, n0 + 16, k0, half, l16, N, K);
      c00 = wmma32(a0, b0, c00);
      c01 = wmma32(a0, b1, c01);
      c10 = wmma32(a1, b0, c10);
      c11 = wmma32(a1, b1, c11);
    }
  }
  store_tile(c00, Cadd, bias, D, ldd, m0,      n0,      M, N, half, l16, act);
  store_tile(c01, Cadd, bias, D, ldd, m0,      n0 + 16, M, N, half, l16, act);
  store_tile(c10, Cadd, bias, D, ldd, m0 + 16, n0,      M, N, half, l16, act);
  store_tile(c11, Cadd, bias, D, ldd, m0 + 16, n0 + 16, M, N, half, l16, act);
}

// ---------------------------------------------------------------- segment ops
__global__ void k_scatter_mean(const float* __restrict__ y, const int* __restrict__ src,
                               const int* __restrict__ dst, int E, int nself, int C,
                               float* __restrict__ acc, float* __restrict__ cnt){
  long long idx = (long long)blockIdx.x*256 + threadIdx.x;
  long long total = (long long)(E + nself)*C;
  if (idx >= total) return;
  int e = (int)(idx / C), c = (int)(idx % C);
  int s, d; edge_sd(e, E, src, dst, s, d);
  atomicAdd(&acc[(size_t)d*C + c], y[(size_t)s*C + c]);
  if (c == 0) atomicAdd(&cnt[d], 1.0f);
}
__global__ void k_finalize_mean(const float* __restrict__ acc, const float* __restrict__ cnt,
                                const float* __restrict__ bias, float* __restrict__ out,
                                long long total, int C, int act){
  long long idx = (long long)blockIdx.x*256 + threadIdx.x;
  if (idx >= total) return;
  int i = (int)(idx / C), c = (int)(idx % C);
  float v = acc[idx] / fmaxf(cnt[i], 1.0f);
  if (bias) v += bias[c];
  if (act == 2) v = v > 0.f ? v : (expf(v) - 1.f);
  else if (act == 1) v = v > 0.f ? v : 0.f;
  out[idx] = v;
}
__global__ void k_colmean(const float* __restrict__ x, int n, int C, float* __restrict__ out){
  int c = blockIdx.x*256 + threadIdx.x;
  if (c >= C) return;
  float s = 0.f;
  for (int i = 0; i < n; ++i) s += x[(size_t)i*C + c];
  out[c] = s / (float)n;
}
__global__ void k_scatter_max_feat(const float* __restrict__ x, const int* __restrict__ src,
                                   const int* __restrict__ dst, int E, int nself, int C,
                                   unsigned* __restrict__ enc){
  long long idx = (long long)blockIdx.x*256 + threadIdx.x;
  long long total = (long long)(E + nself)*C;
  if (idx >= total) return;
  int e = (int)(idx / C), c = (int)(idx % C);
  int s, d; edge_sd(e, E, src, dst, s, d);
  atomicMax(&enc[(size_t)d*C + c], fenc(x[(size_t)s*C + c]));
}
__global__ void k_decode_max(const unsigned* __restrict__ enc, float* __restrict__ out, long long total){
  long long i = (long long)blockIdx.x*256 + threadIdx.x;
  if (i < total) out[i] = fdec(enc[i]);
}
__global__ void k_edge_dot_lrelu(const float* __restrict__ x, const float* __restrict__ xq,
                                 const int* __restrict__ src, const int* __restrict__ dst,
                                 int E, int nself, float* __restrict__ sE){
  int e = blockIdx.x*256 + threadIdx.x;
  if (e >= E + nself) return;
  int s, d; edge_sd(e, E, src, dst, s, d);
  float acc = 0.f;
  for (int c = 0; c < 64; ++c) acc += xq[(size_t)d*64 + c] * x[(size_t)s*64 + c];
  sE[e] = lrelu02(acc);
}
__global__ void k_scatter_max_scalar(const float* __restrict__ sE, const int* __restrict__ src,
                                     const int* __restrict__ dst, int E, int nself,
                                     unsigned* __restrict__ enc){
  int e = blockIdx.x*256 + threadIdx.x;
  if (e >= E + nself) return;
  int s, d; edge_sd(e, E, src, dst, s, d); (void)s;
  atomicMax(&enc[d], fenc(sE[e]));
}
__global__ void k_edge_exp(const float* __restrict__ sE, const unsigned* __restrict__ encM,
                           const int* __restrict__ src, const int* __restrict__ dst,
                           int E, int nself, float* __restrict__ eE, float* __restrict__ denom){
  int e = blockIdx.x*256 + threadIdx.x;
  if (e >= E + nself) return;
  int s, d; edge_sd(e, E, src, dst, s, d); (void)s;
  float v = expf(sE[e] - fdec(encM[d]));
  eE[e] = v;
  atomicAdd(&denom[d], v);
}
__global__ void k_edge_score(const float* __restrict__ eE, const float* __restrict__ denom,
                             const int* __restrict__ src, const int* __restrict__ dst,
                             int E, int nself, float* __restrict__ scE){
  int e = blockIdx.x*256 + threadIdx.x;
  if (e >= E + nself) return;
  int s, d; edge_sd(e, E, src, dst, s, d); (void)s;
  scE[e] = eE[e] / denom[d];
}
__global__ void k_scatter_score_feat(const float* __restrict__ x, const float* __restrict__ scE,
                                     const int* __restrict__ src, const int* __restrict__ dst,
                                     int E, int nself, int C, float* __restrict__ xc){
  long long idx = (long long)blockIdx.x*256 + threadIdx.x;
  long long total = (long long)(E + nself)*C;
  if (idx >= total) return;
  int e = (int)(idx / C), c = (int)(idx % C);
  int s, d; edge_sd(e, E, src, dst, s, d);
  atomicAdd(&xc[(size_t)d*C + c], x[(size_t)s*C + c] * scE[e]);
}
__global__ void k_node_ab(const float* __restrict__ xc, int n,
                          const float* __restrict__ w1, const float* __restrict__ b1p,
                          const float* __restrict__ w2,
                          float* __restrict__ avec, float* __restrict__ bvec){
  int i = blockIdx.x*256 + threadIdx.x;
  if (i >= n) return;
  float a = 0.f, b = 0.f;
  for (int c = 0; c < 64; ++c) {
    float v = xc[(size_t)i*64 + c];
    a += v * w1[c]; b += v * w2[c];
  }
  avec[i] = a + b1p[0];
  bvec[i] = b;
}
__global__ void k_scatter_agg(const float* __restrict__ avec, const float* __restrict__ bvec,
                              const int* __restrict__ src, const int* __restrict__ dst,
                              int E, int nself, float* __restrict__ agg1){
  int e = blockIdx.x*256 + threadIdx.x;
  if (e >= E + nself) return;
  int s, d; edge_sd(e, E, src, dst, s, d);
  atomicAdd(&agg1[d], avec[s] - bvec[d]);
}
__global__ void k_fitness(const float* __restrict__ xc, const float* __restrict__ agg1,
                          const float* __restrict__ w3, const float* __restrict__ b3p,
                          int n, float* __restrict__ fit){
  int i = blockIdx.x*256 + threadIdx.x;
  if (i >= n) return;
  float s = 0.f;
  for (int c = 0; c < 64; ++c) s += xc[(size_t)i*64 + c] * w3[c];
  float z = agg1[i] + s + b3p[0];
  fit[i] = 1.f / (1.f + expf(-z));
}
__global__ void k_topk(const float* __restrict__ fit, int n, int k,
                       int* __restrict__ perm, float* __restrict__ fitk){
  int i = blockIdx.x*256 + threadIdx.x;
  if (i >= n) return;
  float fi = fit[i];
  int r = 0;
  for (int j = 0; j < n; ++j) {
    float fj = fit[j];
    if (fj > fi || (fj == fi && j < i)) ++r;
  }
  if (r < k) { perm[r] = i; fitk[r] = fi; }
}
__global__ void k_gather_scale(const float* __restrict__ xc, const int* __restrict__ perm,
                               const float* __restrict__ fitk, int k, int C, float* __restrict__ out){
  long long idx = (long long)blockIdx.x*256 + threadIdx.x;
  long long total = (long long)k*C;
  if (idx >= total) return;
  int r = (int)(idx / C), c = (int)(idx % C);
  out[idx] = xc[(size_t)perm[r]*C + c] * fitk[r];
}
__global__ void k_scatter_SA(const float* __restrict__ scE, const int* __restrict__ src,
                             const int* __restrict__ dst, int E, int nself, int n,
                             float* __restrict__ S, float* __restrict__ A){
  int e = blockIdx.x*256 + threadIdx.x;
  if (e >= E + nself) return;
  int s, d; edge_sd(e, E, src, dst, s, d);
  atomicAdd(&S[(size_t)s*n + d], scE[e]);
  atomicAdd(&A[(size_t)s*n + d], 1.0f);
}
__global__ void k_gather_cols(const float* __restrict__ S, const int* __restrict__ perm,
                              int n, int k, float* __restrict__ Sp){
  long long idx = (long long)blockIdx.x*256 + threadIdx.x;
  long long total = (long long)n*k;
  if (idx >= total) return;
  int i = (int)(idx / k), r = (int)(idx % k);
  Sp[(size_t)i*k + r] = S[(size_t)i*n + perm[r]];
}
__global__ void k_zero_diag(float* __restrict__ M, int k){
  int r = blockIdx.x*256 + threadIdx.x;
  if (r < k) M[(size_t)r*k + r] = 0.f;
}

// ---------------------------------------------------------------- dense-graph ops
__global__ void k_diag_fix(float* __restrict__ A, int n){
  int i = blockIdx.x*256 + threadIdx.x;
  if (i < n && A[(size_t)i*n + i] == 0.f) A[(size_t)i*n + i] = 1.f;
}
__global__ void k_colmax_masked_x(const float* __restrict__ A, const float* __restrict__ x,
                                  int n, float* __restrict__ xqraw){
  __shared__ float sa[64];
  int j = blockIdx.x;
  int c = threadIdx.x;   // 64 threads
  float v = NEGV;
  for (int i0 = 0; i0 < n; i0 += 64) {
    int lim = n - i0; if (lim > 64) lim = 64;
    __syncthreads();
    if (c < lim) sa[c] = A[(size_t)(i0 + c)*n + j];
    __syncthreads();
    for (int t = 0; t < lim; ++t)
      if (sa[t] != 0.f) v = fmaxf(v, x[(size_t)(i0 + t)*64 + c]);
  }
  xqraw[(size_t)j*64 + c] = v;
}
__global__ void k_mask_lrelu(const float* __restrict__ A, float* __restrict__ G, long long total){
  long long idx = (long long)blockIdx.x*256 + threadIdx.x;
  if (idx >= total) return;
  G[idx] = (A[idx] != 0.f) ? lrelu02(G[idx]) : NEGV;
}
__global__ void k_colmax(const float* __restrict__ Sc, int n, float* __restrict__ m){
  int j = blockIdx.x*256 + threadIdx.x;
  if (j >= n) return;
  float v = -3.0e38f;
  for (int i = 0; i < n; ++i) v = fmaxf(v, Sc[(size_t)i*n + j]);
  m[j] = v;
}
__global__ void k_exp_sub(float* __restrict__ P, const float* __restrict__ m, long long total, int n){
  long long idx = (long long)blockIdx.x*256 + threadIdx.x;
  if (idx >= total) return;
  int j = (int)(idx % n);
  P[idx] = expf(P[idx] - m[j]);   // masked entries: exp(NEG - m) underflows to 0
}
__global__ void k_colsum(const float* __restrict__ P, int n, float* __restrict__ cs){
  int j = blockIdx.x*256 + threadIdx.x;
  if (j >= n) return;
  float s = 0.f;
  for (int i = 0; i < n; ++i) s += P[(size_t)i*n + j];
  cs[j] = s;
}
__global__ void k_divcol(float* __restrict__ P, const float* __restrict__ cs, long long total, int n){
  long long idx = (long long)blockIdx.x*256 + threadIdx.x;
  if (idx >= total) return;
  int j = (int)(idx % n);
  P[idx] = P[idx] / cs[j];
}
__global__ void k_dense_agg(const float* __restrict__ A, const float* __restrict__ avec,
                            const float* __restrict__ bvec, int n, float* __restrict__ agg1){
  int j = blockIdx.x*256 + threadIdx.x;
  if (j >= n) return;
  float sa = 0.f; float cnt = 0.f;
  for (int i = 0; i < n; ++i) {
    float v = A[(size_t)i*n + j];
    if (v != 0.f) { cnt += 1.f; sa += avec[i]; }
  }
  agg1[j] = sa - cnt * bvec[j];
}
__global__ void k_colcnt(const float* __restrict__ A, int n, float* __restrict__ colc){
  int j = blockIdx.x*256 + threadIdx.x;
  if (j >= n) return;
  float cnt = 0.f;
  for (int i = 0; i < n; ++i) if (A[(size_t)i*n + j] != 0.f) cnt += 1.f;
  colc[j] = cnt;
}
__global__ void k_scale_rows(float* __restrict__ xagg, const float* __restrict__ colc, int n, int C){
  long long idx = (long long)blockIdx.x*256 + threadIdx.x;
  long long total = (long long)n*C;
  if (idx >= total) return;
  int j = (int)(idx / C);
  xagg[idx] = xagg[idx] / fmaxf(colc[j], 1.0f);
}

// ---------------------------------------------------------------- classifier
__global__ void k_classifier(const float* __restrict__ xs, const float* __restrict__ linW1,
                             const float* __restrict__ linb1, const float* __restrict__ linW2,
                             const float* __restrict__ linb2, float* __restrict__ out){
  __shared__ float h1[64];
  __shared__ float lg[10];
  int t = threadIdx.x;  // 64 threads
  if (t < 64) {
    float s = linb1[t];
    for (int r = 0; r < 640; ++r) s += xs[r] * linW1[(size_t)r*64 + t];
    h1[t] = s > 0.f ? s : 0.f;
  }
  __syncthreads();
  if (t < 10) {
    float s = linb2[t];
    for (int r = 0; r < 64; ++r) s += h1[r] * linW2[(size_t)r*10 + t];
    lg[t] = s;
  }
  __syncthreads();
  if (t == 0) {
    float mx = lg[0];
    for (int i = 1; i < 10; ++i) mx = fmaxf(mx, lg[i]);
    float se = 0.f;
    for (int i = 0; i < 10; ++i) se += expf(lg[i] - mx);
    float lse = mx + logf(se);
    for (int i = 0; i < 10; ++i) out[i] = lg[i] - lse;
  }
}

// ================================================================ host
extern "C" void kernel_launch(void* const* d_in, const int* in_sizes, int n_in,
                              void* d_out, int out_size, void* d_ws, size_t ws_size,
                              hipStream_t stream)
{
  (void)n_in; (void)out_size; (void)ws_size;
  const float* xin   = (const float*)d_in[0];
  const int*   esrc  = (const int*)d_in[1];
  const int*   edst  = (const int*)d_in[2];
  const float* W1    = (const float*)d_in[3];
  const float* b1    = (const float*)d_in[4];
  const float* W2    = (const float*)d_in[5];
  const float* b2    = (const float*)d_in[6];
  const float* W3    = (const float*)d_in[7];
  const float* b3    = (const float*)d_in[8];
  const float* Wrel  = (const float*)d_in[9];
  const float* brel  = (const float*)d_in[10];
  const float* Wroot = (const float*)d_in[11];
  const float* pW    = (const float*)d_in[12];
  const float* pb    = (const float*)d_in[13];
  const float* leW1  = (const float*)d_in[14];
  const float* leb1  = (const float*)d_in[15];
  const float* leW2  = (const float*)d_in[16];
  const float* leW3  = (const float*)d_in[17];
  const float* leb3  = (const float*)d_in[18];
  const float* linW1 = (const float*)d_in[19];
  const float* linb1 = (const float*)d_in[20];
  const float* linW2 = (const float*)d_in[21];
  const float* linb2 = (const float*)d_in[22];

  const int N0 = in_sizes[0] / 16;
  const int E  = in_sizes[1];
  const int EL = E + N0;

  // ---- workspace carve
  char* wp = (char*)d_ws;
  auto carve = [&](size_t bytes)->void*{
    void* p = (void*)wp; wp += (bytes + 255) & ~(size_t)255; return p; };
  size_t NN = (size_t)N0 * (size_t)N0;
  float* bigs[4];
  for (int i = 0; i < 4; ++i) bigs[i] = (float*)carve(NN * sizeof(float));
  float* fbuf[2];
  fbuf[0] = (float*)carve((size_t)N0 * 64 * sizeof(float));
  fbuf[1] = (float*)carve((size_t)N0 * 64 * sizeof(float));
  float* ftmp   = (float*)carve((size_t)N0 * 64 * sizeof(float));
  float* fagg   = (float*)carve((size_t)N0 * 64 * sizeof(float));
  float* fxq    = (float*)carve((size_t)N0 * 64 * sizeof(float));
  float* fxc    = (float*)carve((size_t)N0 * 64 * sizeof(float));
  float* fxqraw = (float*)carve((size_t)N0 * 64 * sizeof(float));
  float* sE  = (float*)carve((size_t)EL * sizeof(float));
  float* eE  = (float*)carve((size_t)EL * sizeof(float));
  float* scE = (float*)carve((size_t)EL * sizeof(float));
  float* mbuf  = (float*)carve((size_t)N0 * sizeof(float));
  float* denom = (float*)carve((size_t)N0 * sizeof(float));
  float* cntf  = (float*)carve((size_t)N0 * sizeof(float));
  float* agg1  = (float*)carve((size_t)N0 * sizeof(float));
  float* avec  = (float*)carve((size_t)N0 * sizeof(float));
  float* bvec  = (float*)carve((size_t)N0 * sizeof(float));
  float* fit   = (float*)carve((size_t)N0 * sizeof(float));
  float* fitk  = (float*)carve((size_t)N0 * sizeof(float));
  float* colc  = (float*)carve((size_t)N0 * sizeof(float));
  unsigned* encNC = (unsigned*)carve((size_t)N0 * 64 * sizeof(unsigned));
  unsigned* encN  = (unsigned*)carve((size_t)N0 * sizeof(unsigned));
  int* perm = (int*)carve((size_t)N0 * sizeof(int));
  float* xsb = (float*)carve(640 * sizeof(float));

  auto blocks = [](long long t)->dim3{ return dim3((unsigned)((t + 255) / 256)); };
  auto fillf = [&](float* p, long long c, float v){ k_fill_f32<<<blocks(c),256,0,stream>>>(p,c,v); };
  auto fillu = [&](unsigned* p, long long c, unsigned v){ k_fill_u32<<<blocks(c),256,0,stream>>>(p,c,v); };
  auto gemm = [&](const float* A,int lda,int tA,const float* B,int ldb,int tB,
                  const float* Cadd,const float* bias,float* D,int ldd,
                  int M,int Nn,int K,int act){
    dim3 g((Nn + 31)/32, (M + 31)/32);
    if (tA == 0 && tB == 0)
      k_wmma_gemm_t<0,0><<<g,32,0,stream>>>(A,lda,B,ldb,Cadd,bias,D,ldd,M,Nn,K,act);
    else if (tA == 1 && tB == 0)
      k_wmma_gemm_t<1,0><<<g,32,0,stream>>>(A,lda,B,ldb,Cadd,bias,D,ldd,M,Nn,K,act);
    else if (tA == 0 && tB == 1)
      k_wmma_gemm_t<0,1><<<g,32,0,stream>>>(A,lda,B,ldb,Cadd,bias,D,ldd,M,Nn,K,act);
    else
      k_wmma_gemm_t<1,1><<<g,32,0,stream>>>(A,lda,B,ldb,Cadd,bias,D,ldd,M,Nn,K,act);
  };

  // feature double-buffer
  const float* cur = xin;
  int wi = 0;
  auto commit = [&](){ cur = fbuf[wi]; wi ^= 1; };

  int n = N0;

  // ---- FeaSt layer: out = elu(seg_mean((x@W)[srcL], dstL) + b)
  auto feast = [&](const float* X, int Cin, int Cout, const float* W, const float* b){
    gemm(X, Cin, 0, W, Cout, 0, nullptr, nullptr, ftmp, Cout, n, Cout, Cin, 0);
    fillf(fagg, (long long)n*Cout, 0.f);
    fillf(cntf, n, 0.f);
    k_scatter_mean<<<blocks((long long)(E+n)*Cout),256,0,stream>>>(ftmp, esrc, edst, E, n, Cout, fagg, cntf);
    k_finalize_mean<<<blocks((long long)n*Cout),256,0,stream>>>(fagg, cntf, b, fbuf[wi], (long long)n*Cout, Cout, 2);
    commit();
  };

  feast(cur, 16, 32, W1, b1);
  feast(cur, 32, 64, W2, b2);
  feast(cur, 64, 64, W3, b3);
  k_colmean<<<1,256,0,stream>>>(cur, n, 64, xsb + 0);

  int aIdx = -1;   // -1: sparse phase (no dense A yet)

  for (int i = 0; i < 9; ++i) {
    const float* Wr = Wrel + (size_t)i*64*64;
    const float* br = brel + (size_t)i*64;
    const float* Wo = Wroot + (size_t)i*64*64;

    if (aIdx < 0) {
      // gconv_sparse: relu(seg_mean(x[src],dst)@Wr + br + x@Wo)
      fillf(fagg, (long long)n*64, 0.f);
      fillf(cntf, n, 0.f);
      k_scatter_mean<<<blocks((long long)E*64),256,0,stream>>>(cur, esrc, edst, E, 0, 64, fagg, cntf);
      k_finalize_mean<<<blocks((long long)n*64),256,0,stream>>>(fagg, cntf, nullptr, fxqraw, (long long)n*64, 64, 0);
      gemm(cur, 64, 0, Wo, 64, 0, nullptr, nullptr, ftmp, 64, n, 64, 64, 0);
      gemm(fxqraw, 64, 0, Wr, 64, 0, ftmp, br, fbuf[wi], 64, n, 64, 64, 1);
      commit();
    } else {
      // gconv_dense
      float* A = bigs[aIdx];
      k_colcnt<<<blocks(n),256,0,stream>>>(A, n, colc);
      gemm(A, n, 1, cur, 64, 0, nullptr, nullptr, fagg, 64, n, 64, n, 0);   // agg = A^T @ x
      k_scale_rows<<<blocks((long long)n*64),256,0,stream>>>(fagg, colc, n, 64);
      gemm(cur, 64, 0, Wo, 64, 0, nullptr, nullptr, ftmp, 64, n, 64, 64, 0);
      gemm(fagg, 64, 0, Wr, 64, 0, ftmp, br, fbuf[wi], 64, n, 64, 64, 1);
      commit();
    }
    k_colmean<<<1,256,0,stream>>>(cur, n, 64, xsb + (size_t)(i+1)*64);

    if (i % 2 == 0 && i < 8) {
      int p = i / 2;
      int k = (9*n + 9) / 10;   // ceil(0.9*n)
      const float* pWp = pW + (size_t)p*64*64;
      const float* pbp = pb + (size_t)p*64;
      const float* w1p = leW1 + (size_t)p*64;
      const float* b1p = leb1 + p;
      const float* w2p = leW2 + (size_t)p*64;
      const float* w3p = leW3 + (size_t)p*64;
      const float* b3p = leb3 + p;

      if (aIdx < 0) {
        // ---------- ASAP sparse (n = N0) ----------
        fillu(encNC, (long long)n*64, ENCNEG);
        k_scatter_max_feat<<<blocks((long long)(E+n)*64),256,0,stream>>>(cur, esrc, edst, E, n, 64, encNC);
        k_decode_max<<<blocks((long long)n*64),256,0,stream>>>(encNC, fxqraw, (long long)n*64);
        gemm(fxqraw, 64, 0, pWp, 64, 0, nullptr, pbp, fxq, 64, n, 64, 64, 0);
        k_edge_dot_lrelu<<<blocks(E+n),256,0,stream>>>(cur, fxq, esrc, edst, E, n, sE);
        fillu(encN, n, ENCNEG);
        k_scatter_max_scalar<<<blocks(E+n),256,0,stream>>>(sE, esrc, edst, E, n, encN);
        fillf(denom, n, 0.f);
        k_edge_exp<<<blocks(E+n),256,0,stream>>>(sE, encN, esrc, edst, E, n, eE, denom);
        k_edge_score<<<blocks(E+n),256,0,stream>>>(eE, denom, esrc, edst, E, n, scE);
        fillf(fxc, (long long)n*64, 0.f);
        k_scatter_score_feat<<<blocks((long long)(E+n)*64),256,0,stream>>>(cur, scE, esrc, edst, E, n, 64, fxc);
        k_node_ab<<<blocks(n),256,0,stream>>>(fxc, n, w1p, b1p, w2p, avec, bvec);
        fillf(agg1, n, 0.f);
        k_scatter_agg<<<blocks(E+n),256,0,stream>>>(avec, bvec, esrc, edst, E, n, agg1);
        k_fitness<<<blocks(n),256,0,stream>>>(fxc, agg1, w3p, b3p, n, fit);
        k_topk<<<blocks(n),256,0,stream>>>(fit, n, k, perm, fitk);
        k_gather_scale<<<blocks((long long)k*64),256,0,stream>>>(fxc, perm, fitk, k, 64, fbuf[wi]);
        // S = bigs[0], A = bigs[1], Sp = bigs[2], T = bigs[3], Anew -> bigs[0]
        fillf(bigs[0], (long long)NN, 0.f);
        fillf(bigs[1], (long long)NN, 0.f);
        k_scatter_SA<<<blocks(E+n),256,0,stream>>>(scE, esrc, edst, E, n, n, bigs[0], bigs[1]);
        k_gather_cols<<<blocks((long long)n*k),256,0,stream>>>(bigs[0], perm, n, k, bigs[2]);
        gemm(bigs[1], n, 0, bigs[2], k, 0, nullptr, nullptr, bigs[3], k, n, k, n, 0);   // T = A @ Sp
        gemm(bigs[2], k, 1, bigs[3], k, 0, nullptr, nullptr, bigs[0], k, k, k, n, 0);   // Anew = Sp^T @ T
        k_zero_diag<<<blocks(k),256,0,stream>>>(bigs[0], k);
        aIdx = 0;
        commit();
        n = k;
      } else {
        // ---------- ASAP dense ----------
        float* A = bigs[aIdx];
        int o[3]; { int t = 0; for (int q = 0; q < 4; ++q) if (q != aIdx) o[t++] = q; }
        float* P  = bigs[o[0]];  // Sc / e / score, then Anew
        float* Sp = bigs[o[1]];
        float* T  = bigs[o[2]];
        long long nn2 = (long long)n*n;
        k_diag_fix<<<blocks(n),256,0,stream>>>(A, n);
        k_colmax_masked_x<<<n,64,0,stream>>>(A, cur, n, fxqraw);
        gemm(fxqraw, 64, 0, pWp, 64, 0, nullptr, pbp, fxq, 64, n, 64, 64, 0);
        gemm(cur, 64, 0, fxq, 64, 1, nullptr, nullptr, P, n, n, n, 64, 0);  // G = x @ xq^T
        k_mask_lrelu<<<blocks(nn2),256,0,stream>>>(A, P, nn2);
        k_colmax<<<blocks(n),256,0,stream>>>(P, n, mbuf);
        k_exp_sub<<<blocks(nn2),256,0,stream>>>(P, mbuf, nn2, n);
        k_colsum<<<blocks(n),256,0,stream>>>(P, n, denom);
        k_divcol<<<blocks(nn2),256,0,stream>>>(P, denom, nn2, n);           // P = score
        gemm(P, n, 1, cur, 64, 0, nullptr, nullptr, fxc, 64, n, 64, n, 0);  // xc = score^T @ x
        k_node_ab<<<blocks(n),256,0,stream>>>(fxc, n, w1p, b1p, w2p, avec, bvec);
        k_dense_agg<<<blocks(n),256,0,stream>>>(A, avec, bvec, n, agg1);
        k_fitness<<<blocks(n),256,0,stream>>>(fxc, agg1, w3p, b3p, n, fit);
        k_topk<<<blocks(n),256,0,stream>>>(fit, n, k, perm, fitk);
        k_gather_scale<<<blocks((long long)k*64),256,0,stream>>>(fxc, perm, fitk, k, 64, fbuf[wi]);
        k_gather_cols<<<blocks((long long)n*k),256,0,stream>>>(P, perm, n, k, Sp);
        gemm(A, n, 0, Sp, k, 0, nullptr, nullptr, T, k, n, k, n, 0);        // T = A @ Sp
        gemm(Sp, k, 1, T, k, 0, nullptr, nullptr, P, k, k, k, n, 0);        // Anew = Sp^T @ T
        k_zero_diag<<<blocks(k),256,0,stream>>>(P, k);
        aIdx = o[0];
        commit();
        n = k;
      }
    }
  }

  k_classifier<<<1,64,0,stream>>>(xsb, linW1, linb1, linW2, linb2, (float*)d_out);
}
